// SwinDADSModel_13391708029558
// MI455X (gfx1250) — matrix operand order repaired and checked by reading, not compiled, not run on hardware
//
#include <hip/hip_runtime.h>

typedef __attribute__((ext_vector_type(16))) _Float16 v16h;
typedef __attribute__((ext_vector_type(8)))  float    v8f;
typedef __attribute__((ext_vector_type(4)))  float    f32x4;
typedef __attribute__((ext_vector_type(4)))  unsigned int u32x4;
typedef __attribute__((ext_vector_type(8)))  int      i32x8;
typedef __attribute__((ext_vector_type(4)))  int      i32x4;

#define WMMA_F32_F16(a, b, c) \
  __builtin_amdgcn_wmma_f32_16x16x32_f16(false, (a), false, (b), (short)0, (c), false, false)

// ---------------------------------------------------------------------------
// Weight prep: W (N,K) f32 -> Wt (K,N) f16.  (K,N) layout makes every 32x64
// B-tile a dense 2D tile with row stride N -> ideal for the Tensor Data Mover.
// ---------------------------------------------------------------------------
__global__ void cast_transpose_kernel(const float* __restrict__ w,
                                      _Float16* __restrict__ wt, int N, int K) {
  size_t idx = (size_t)blockIdx.x * blockDim.x + threadIdx.x;
  if (idx >= (size_t)N * K) return;
  int n = (int)(idx % N);
  int k = (int)(idx / N);
  wt[idx] = (_Float16)w[(size_t)n * K + k];
}

// ---------------------------------------------------------------------------
// Generic WMMA GEMM: C[M,N] = A[M,K](f32) * Wt[K,N](f16) (+bias)(+gelu|+resid)
// Tile 64x64, K-step 32, 8 waves (2 M x 4 N).
//  - A tile: vectorized float4 loads, packed f32->f16 convert, b64 LDS stores
//  - B tile: Tensor Data Mover (tensor_load_to_lds) issued by wave 0,
//            completion via s_wait_tensorcnt, published by the block barrier.
//            tensor_dim0 = N-bn gives hardware zero-fill at the N edge.
// mode: 0 = bias only, 1 = bias + exact GELU, 2 = bias + residual add
// ---------------------------------------------------------------------------
__global__ __launch_bounds__(256) void gemm_wmma_kernel(
    const float* __restrict__ A, const _Float16* __restrict__ Wt,
    const float* __restrict__ bias, const float* __restrict__ resid,
    float* __restrict__ C, int M, int N, int K, int mode) {
  __shared__ _Float16 As[64][40];   // 64 x 32 (padded)
  __shared__ _Float16 Bs[32][64];   // 32 x 64, contiguous (TDM destination)

  const int tid   = threadIdx.x;
  const int lane  = tid & 31;
  const int wave  = tid >> 5;
  const int lmod  = lane & 15;
  const int lhalf = lane >> 4;
  const int wm    = wave >> 2;      // 0..1 -> 32-row half
  const int wn    = wave & 3;       // 0..3 -> 16-col strip
  const int bm    = blockIdx.x * 64;
  const int bn    = blockIdx.y * 64;

  const unsigned int ldsBs = (unsigned int)(uintptr_t)(&Bs[0][0]); // LDS byte offset
  const int td0 = N - bn;           // valid columns in this tile (OOB cols -> 0)

  v8f acc0 = {}; v8f acc1 = {};

  for (int k0 = 0; k0 < K; k0 += 32) {
    // ---- A tile: 64x32 f32, two float4 loads per thread ----
    for (int i = tid; i < 64 * 8; i += 256) {
      int r = i >> 3, c4 = (i & 7) * 4;
      int gr = bm + r;
      f32x4 v = {0.f, 0.f, 0.f, 0.f};
      if (gr < M) {
        const float* ap = A + (size_t)gr * K + k0 + c4;
        v = *(const f32x4*)ap;
        if (k0 + 32 < K) __builtin_prefetch(ap + 32, 0, 1);  // next K-slice
      }
      As[r][c4 + 0] = (_Float16)v.x;
      As[r][c4 + 1] = (_Float16)v.y;
      As[r][c4 + 2] = (_Float16)v.z;
      As[r][c4 + 3] = (_Float16)v.w;
    }

    // ---- B tile via Tensor Data Mover (one wave issues the DMA) ----
    if (wave == 0) {
#if defined(__HIP_DEVICE_COMPILE__)
      unsigned long long ga =
          (unsigned long long)(uintptr_t)(Wt + (size_t)k0 * N + bn);
      // D# group 0: count=1 | lds_addr | global_addr[56:0] | type=2
      u32x4 g0 = {1u, ldsBs, (unsigned int)ga,
                  (unsigned int)((ga >> 32) & 0x01ffffffu) | 0x80000000u};
      // D# group 1: data_size=2B; tensor_dim0=td0; tensor_dim1=32;
      //             tile_dim0=64; tile_dim1=32; tensor_dim0_stride=N
      i32x8 g1 = {(int)(1u << 16),
                  (int)(((unsigned)td0 & 0xffffu) << 16),
                  (int)((((unsigned)td0 >> 16) & 0xffffu) | (32u << 16)),
                  (int)(64u << 16),
                  (int)32,
                  (int)N,
                  0, 0};
      i32x4 z4 = {0, 0, 0, 0};
      i32x8 z8 = {0, 0, 0, 0, 0, 0, 0, 0};
      __builtin_amdgcn_tensor_load_to_lds(g0, g1, z4, z4, z8, 0);
      __builtin_amdgcn_s_wait_tensorcnt(0);
#endif
    }
    __syncthreads();

    // ---- fragments + WMMA ----
    v16h a0, a1, bf;
#pragma unroll
    for (int e = 0; e < 16; ++e) {
      int kk = e + (e & 8) + lhalf * 8;            // ISA A 16x32 f16 layout
      a0[e] = As[wm * 32 + lmod][kk];
      a1[e] = As[wm * 32 + 16 + lmod][kk];
    }
#pragma unroll
    for (int e = 0; e < 16; ++e)
      bf[e] = Bs[e + (e & 8) + lhalf * 8][wn * 16 + lmod];  // ISA B 32x16 f16 layout

    acc0 = WMMA_F32_F16(a0, bf, acc0);
    acc1 = WMMA_F32_F16(a1, bf, acc1);
    __syncthreads();
  }

  // Epilogue.  C/D layout: row = 8*lhalf + v (tile), col = lmod.
  int col = bn + wn * 16 + lmod;
  if (col < N) {
    float bv = bias ? bias[col] : 0.0f;
#pragma unroll
    for (int g = 0; g < 2; ++g) {
      v8f acc = g ? acc1 : acc0;
#pragma unroll
      for (int v = 0; v < 8; ++v) {
        int r = bm + wm * 32 + g * 16 + lhalf * 8 + v;
        if (r < M) {
          float o = acc[v] + bv;
          if (mode == 1) o = 0.5f * o * (1.0f + erff(o * 0.7071067811865476f));
          else if (mode == 2) o += resid[(size_t)r * N + col];
          C[(size_t)r * N + col] = o;
        }
      }
    }
  }
}

// ---------------------------------------------------------------------------
// Window attention: one workgroup (4 waves) per (window, head).
// 49 tokens padded to 64, head_dim = 32.  S=QK^T and O=PV via WMMA.
// ---------------------------------------------------------------------------
__global__ __launch_bounds__(128) void win_attn_kernel(
    const float* __restrict__ qkv, const float* __restrict__ rpb,
    float* __restrict__ out, int B, int R, int heads, int C, int shift) {
  const int nW = R / 7;
  const int wid = blockIdx.x;
  const int head = blockIdx.y;
  const int b  = wid / (nW * nW);
  const int w  = wid % (nW * nW);
  const int wh = w / nW, ww = w % nW;

  __shared__ _Float16 Qs[64][32];
  __shared__ _Float16 Ks[64][32];
  __shared__ _Float16 Vs[64][32];
  __shared__ float    Sb[64][64];
  __shared__ _Float16 Pb[64][64];

  const int tid   = threadIdx.x;
  const int lane  = tid & 31;
  const int wave  = tid >> 5;
  const int lmod  = lane & 15;
  const int lhalf = lane >> 4;
  const int C3    = 3 * C;
  const float scale = 0.17677669529663687f;   // 32^-0.5

  for (int i = tid; i < 64 * 32; i += 128) {
    int p = i >> 5, c = i & 31;
    float q = 0.f, k = 0.f, v = 0.f;
    if (p < 49) {
      int pi = p / 7, pj = p % 7;
      int sh = (wh * 7 + pi + shift) % R;
      int sw = (ww * 7 + pj + shift) % R;
      const float* base = qkv + ((size_t)(b * R + sh) * R + sw) * C3 + head * 32 + c;
      q = base[0]; k = base[C]; v = base[2 * C];
    }
    Qs[p][c] = (_Float16)(q * scale);
    Ks[p][c] = (_Float16)k;
    Vs[p][c] = (_Float16)v;
  }
  __syncthreads();

  // S = Q * K^T : M=64, N=64, K=32 (single WMMA step per tile).
  {
    v16h a, bf;
#pragma unroll
    for (int e = 0; e < 16; ++e)
      a[e] = Qs[wave * 16 + lmod][e + (e & 8) + lhalf * 8];
#pragma unroll
    for (int nt = 0; nt < 4; ++nt) {
#pragma unroll
      for (int e = 0; e < 16; ++e)
        bf[e] = Ks[nt * 16 + lmod][e + (e & 8) + lhalf * 8];   // B[k][n] = K[n][k]
      v8f acc = {};
      acc = WMMA_F32_F16(a, bf, acc);
#pragma unroll
      for (int v = 0; v < 8; ++v)
        Sb[wave * 16 + lhalf * 8 + v][nt * 16 + lmod] = acc[v];
    }
  }
  __syncthreads();

  // Softmax with inline relative-position bias + shift mask (one thread/row).
  if (tid < 64) {
    int i = tid;
    if (i < 49) {
      int pi = i / 7, pj = i % 7;
      int idI = 0;
      if (shift > 0) {
        int hI = wh * 7 + pi, wI = ww * 7 + pj;
        int rh = (hI < R - 7) ? 0 : ((hI < R - shift) ? 1 : 2);
        int rw = (wI < R - 7) ? 0 : ((wI < R - shift) ? 1 : 2);
        idI = rh * 3 + rw;
      }
      float row[49];
      float m = -1e30f;
      for (int j = 0; j < 49; ++j) {
        int qi = j / 7, qj = j % 7;
        int rel = (pi - qi + 6) * 13 + (pj - qj + 6);
        float s = Sb[i][j] + rpb[rel * heads + head];
        if (shift > 0) {
          int hJ = wh * 7 + qi, wJ = ww * 7 + qj;
          int rh = (hJ < R - 7) ? 0 : ((hJ < R - shift) ? 1 : 2);
          int rw = (wJ < R - 7) ? 0 : ((wJ < R - shift) ? 1 : 2);
          if (rh * 3 + rw != idI) s -= 100.0f;
        }
        row[j] = s;
        m = fmaxf(m, s);
      }
      float sum = 0.f;
      for (int j = 0; j < 49; ++j) { row[j] = __expf(row[j] - m); sum += row[j]; }
      float inv = 1.0f / sum;
      for (int j = 0; j < 49; ++j) Pb[i][j] = (_Float16)(row[j] * inv);
      for (int j = 49; j < 64; ++j) Pb[i][j] = (_Float16)0.0f;
    } else {
      for (int j = 0; j < 64; ++j) Pb[i][j] = (_Float16)0.0f;
    }
  }
  __syncthreads();

  // O = P * V : M=64, N=32 (2 tiles), K=64 (2 WMMA steps).
  {
    v8f acc[2] = {};
#pragma unroll
    for (int k0 = 0; k0 < 64; k0 += 32) {
      v16h a;
#pragma unroll
      for (int e = 0; e < 16; ++e)
        a[e] = Pb[wave * 16 + lmod][k0 + e + (e & 8) + lhalf * 8];
#pragma unroll
      for (int nt = 0; nt < 2; ++nt) {
        v16h bf;
#pragma unroll
        for (int e = 0; e < 16; ++e)
          bf[e] = Vs[k0 + e + (e & 8) + lhalf * 8][nt * 16 + lmod];
        acc[nt] = WMMA_F32_F16(a, bf, acc[nt]);
      }
    }
#pragma unroll
    for (int nt = 0; nt < 2; ++nt)
#pragma unroll
      for (int v = 0; v < 8; ++v) {
        int r = wave * 16 + lhalf * 8 + v;
        if (r < 49) {
          int pi = r / 7, pj = r % 7;
          int sh = (wh * 7 + pi + shift) % R;
          int sw = (ww * 7 + pj + shift) % R;
          out[((size_t)(b * R + sh) * R + sw) * C + head * 32 + nt * 16 + lmod] = acc[nt][v];
        }
      }
  }
}

// ---------------------------------------------------------------------------
// LayerNorm over channel dim, one wave (32 lanes) per token.
// ---------------------------------------------------------------------------
__global__ __launch_bounds__(128) void ln_kernel(
    const float* __restrict__ x, const float* __restrict__ g,
    const float* __restrict__ bta, float* __restrict__ y, int tokens, int C) {
  int tok = blockIdx.x * 4 + (threadIdx.x >> 5);
  int lane = threadIdx.x & 31;
  if (tok >= tokens) return;
  const float* xp = x + (size_t)tok * C;
  float sum = 0.f, sq = 0.f;
  for (int c = lane; c < C; c += 32) { float v = xp[c]; sum += v; sq += v * v; }
#pragma unroll
  for (int off = 16; off > 0; off >>= 1) {
    sum += __shfl_xor(sum, off, 32);
    sq  += __shfl_xor(sq,  off, 32);
  }
  float mu = sum / C;
  float var = sq / C - mu * mu;
  float rs = rsqrtf(var + 1e-5f);
  float* yp = y + (size_t)tok * C;
  for (int c = lane; c < C; c += 32)
    yp[c] = (xp[c] - mu) * rs * g[c] + bta[c];
}

// ---------------------------------------------------------------------------
// Patch embed: 4x4 stride-4 conv, 3->96 (K=48, VALU), NCHW -> NHWC.
// ---------------------------------------------------------------------------
__global__ void patch_embed_kernel(const float* __restrict__ x,
                                   const float* __restrict__ w,
                                   const float* __restrict__ b,
                                   float* __restrict__ y) {
  int idx = blockIdx.x * blockDim.x + threadIdx.x;
  if (idx >= 32 * 56 * 56 * 96) return;
  int o = idx % 96;
  int t = idx / 96;
  int pw = t % 56; t /= 56;
  int ph = t % 56;
  int bb = t / 56;
  float acc = b[o];
  for (int c = 0; c < 3; ++c)
    for (int kh = 0; kh < 4; ++kh)
      for (int kw = 0; kw < 4; ++kw)
        acc += x[(((size_t)bb * 3 + c) * 224 + (ph * 4 + kh)) * 224 + (pw * 4 + kw)] *
               w[((o * 3 + c) * 4 + kh) * 4 + kw];
  y[idx] = acc;
}

// ---------------------------------------------------------------------------
// Patch-merge 2x2 gather: (B,R,R,C) -> (B,R/2,R/2,4C), channels x0|x1|x2|x3
// ---------------------------------------------------------------------------
__global__ void merge_gather_kernel(const float* __restrict__ x,
                                    float* __restrict__ y, int B, int R, int C) {
  int R2 = R / 2;
  size_t total = (size_t)B * R2 * R2 * 4 * C;
  size_t idx = (size_t)blockIdx.x * blockDim.x + threadIdx.x;
  if (idx >= total) return;
  int c4 = (int)(idx % (4 * C));
  size_t t = idx / (4 * C);
  int wc = (int)(t % R2); t /= R2;
  int hc = (int)(t % R2);
  int bb = (int)(t / R2);
  int quad = c4 / C, c = c4 % C;
  int dh = (quad == 1 || quad == 3) ? 1 : 0;
  int dw = (quad >= 2) ? 1 : 0;
  y[idx] = x[(((size_t)bb * R + (hc * 2 + dh)) * R + (wc * 2 + dw)) * C + c];
}

// ---------------------------------------------------------------------------
// Global average pool over the 49 final tokens: (32,49,768) -> (32,768)
// ---------------------------------------------------------------------------
__global__ void pool_kernel(const float* __restrict__ x, float* __restrict__ y) {
  int idx = blockIdx.x * blockDim.x + threadIdx.x;
  if (idx >= 32 * 768) return;
  int c = idx % 768, b = idx / 768;
  float s = 0.f;
  for (int t = 0; t < 49; ++t) s += x[((size_t)b * 49 + t) * 768 + c];
  y[idx] = s * (1.0f / 49.0f);
}

// ---------------------------------------------------------------------------
// Host orchestration
// ---------------------------------------------------------------------------
extern "C" void kernel_launch(void* const* d_in, const int* in_sizes, int n_in,
                              void* d_out, int out_size, void* d_ws, size_t ws_size,
                              hipStream_t stream) {
  (void)n_in; (void)out_size; (void)ws_size; (void)in_sizes;
  const int DIMS[4]   = {96, 192, 384, 768};
  const int DEPTHS[4] = {2, 2, 6, 2};
  const int NHEADS[4] = {3, 6, 12, 24};
  const int RES[4]    = {56, 28, 14, 7};
  const int Bn = 32;

  char* ws = (char*)d_ws;
  size_t off = 0;
  auto alloc = [&](size_t bytes) -> void* {
    void* p = ws + off;
    off += (bytes + 255) & ~(size_t)255;
    return p;
  };

  const size_t maxTok = (size_t)32 * 56 * 56;   // 100352
  float* X    = (float*)alloc(maxTok * 96  * sizeof(float));
  float* XN   = (float*)alloc(maxTok * 96  * sizeof(float));
  float* QKV  = (float*)alloc(maxTok * 288 * sizeof(float));
  float* ATT  = (float*)alloc(maxTok * 96  * sizeof(float));
  float* MLP  = (float*)alloc(maxTok * 384 * sizeof(float));
  float* MRG  = (float*)alloc(maxTok * 96  * sizeof(float));
  float* POOL = (float*)alloc(32 * 768 * sizeof(float));

  // cast + transpose weight (N,K) f32 -> (K,N) f16 into workspace
  auto castW = [&](int i, int N, int K) -> const _Float16* {
    size_t n = (size_t)N * K;
    _Float16* d = (_Float16*)alloc(n * sizeof(_Float16));
    cast_transpose_kernel<<<(unsigned)((n + 255) / 256), 256, 0, stream>>>(
        (const float*)d_in[i], d, N, K);
    return d;
  };
  auto gemm = [&](const float* A, const _Float16* W, const float* bias,
                  const float* resid, float* Cout, int M, int N, int K, int mode) {
    dim3 grid((M + 63) / 64, (N + 63) / 64);
    gemm_wmma_kernel<<<grid, 256, 0, stream>>>(A, W, bias, resid, Cout, M, N, K, mode);
  };
  auto ln = [&](const float* x, const float* g, const float* b, float* y,
                int tokens, int C) {
    ln_kernel<<<(tokens + 3) / 4, 128, 0, stream>>>(x, g, b, y, tokens, C);
  };

  // --- Patch embed + LN ---
  {
    int total = 32 * 56 * 56 * 96;
    patch_embed_kernel<<<(total + 255) / 256, 256, 0, stream>>>(
        (const float*)d_in[0], (const float*)d_in[1], (const float*)d_in[2], XN);
    ln(XN, (const float*)d_in[3], (const float*)d_in[4], X, 32 * 56 * 56, 96);
  }

  // --- Stages ---
  int bi = 0;
  for (int s = 0; s < 4; ++s) {
    int C = DIMS[s], heads = NHEADS[s], R = RES[s];
    int tokens = Bn * R * R;
    for (int d = 0; d < DEPTHS[s]; ++d) {
      int base = 5 + bi * 13;
      const float*    ln1g  = (const float*)d_in[base + 0];
      const float*    ln1b  = (const float*)d_in[base + 1];
      const _Float16* qkvW  = castW(base + 2, 3 * C, C);
      const float*    qkvB  = (const float*)d_in[base + 3];
      const _Float16* projW = castW(base + 4, C, C);
      const float*    projB = (const float*)d_in[base + 5];
      const float*    rpb   = (const float*)d_in[base + 6];
      const float*    ln2g  = (const float*)d_in[base + 7];
      const float*    ln2b  = (const float*)d_in[base + 8];
      const _Float16* fc1W  = castW(base + 9, 4 * C, C);
      const float*    fc1B  = (const float*)d_in[base + 10];
      const _Float16* fc2W  = castW(base + 11, C, 4 * C);
      const float*    fc2B  = (const float*)d_in[base + 12];
      int shift = (d % 2 == 1 && R > 7) ? 3 : 0;

      ln(X, ln1g, ln1b, XN, tokens, C);
      gemm(XN, qkvW, qkvB, nullptr, QKV, tokens, 3 * C, C, 0);
      {
        int nWimg = (R / 7) * (R / 7);
        dim3 grid(Bn * nWimg, heads);
        win_attn_kernel<<<grid, 128, 0, stream>>>(QKV, rpb, ATT, Bn, R, heads, C, shift);
      }
      gemm(ATT, projW, projB, X, X, tokens, C, C, 2);            // + residual
      ln(X, ln2g, ln2b, XN, tokens, C);
      gemm(XN, fc1W, fc1B, nullptr, MLP, tokens, 4 * C, C, 1);   // + GELU
      gemm(MLP, fc2W, fc2B, X, X, tokens, C, 4 * C, 2);          // + residual
      ++bi;
    }
    if (s < 3) {
      int mb = 161 + s * 3;
      const float*    mg   = (const float*)d_in[mb + 0];
      const float*    mbt  = (const float*)d_in[mb + 1];
      const _Float16* redW = castW(mb + 2, 2 * C, 4 * C);
      int R2 = R / 2, tokens2 = Bn * R2 * R2;
      size_t total = (size_t)tokens2 * 4 * C;
      merge_gather_kernel<<<(unsigned)((total + 255) / 256), 256, 0, stream>>>(
          X, MRG, Bn, R, C);
      ln(MRG, mg, mbt, ATT, tokens2, 4 * C);
      gemm(ATT, redW, nullptr, nullptr, X, tokens2, 2 * C, 4 * C, 0);
    }
  }

  // --- Head: LN, mean pool, classifier ---
  {
    int tokens = Bn * 49;
    ln(X, (const float*)d_in[170], (const float*)d_in[171], XN, tokens, 768);
    pool_kernel<<<(32 * 768 + 255) / 256, 256, 0, stream>>>(XN, POOL);
    const _Float16* hw = castW(172, 1000, 768);
    gemm(POOL, hw, (const float*)d_in[173], nullptr, (float*)d_out, 32, 1000, 768, 0);
  }
}
